// GATWithMultiHead_24833500905998
// MI455X (gfx1250) — matrix operand order repaired and checked by reading, not compiled, not run on hardware
//
#include <hip/hip_runtime.h>
#include <hip/hip_bf16.h>

typedef __attribute__((ext_vector_type(16))) __bf16        v16bf;
typedef __attribute__((ext_vector_type(8)))  float         v8f;
typedef __attribute__((ext_vector_type(4)))  unsigned int  v4u;
typedef __attribute__((ext_vector_type(8)))  unsigned int  v8u;

#define NN    8192
#define EE    262144
#define EP    (EE + NN)       // edges incl self loops = 270336
#define EP4   (EP * 4)        // edge*head scores      = 1081344
#define HEADS 4
#define FOUT  64
#define CH    256             // HEADS*FOUT

// float -> bf16 bits, round-to-nearest-even (pure integer ops)
__device__ __forceinline__ unsigned f2bf_bits(float f) {
  union { float f; unsigned u; } v; v.f = f;
  return (v.u + 0x7FFFu + ((v.u >> 16) & 1u)) >> 16;
}
__device__ __forceinline__ unsigned pack_bf2(float lo, float hi) {
  return f2bf_bits(lo) | (f2bf_bits(hi) << 16);
}
__device__ __forceinline__ __bf16 f2bf(float f) {
  unsigned short h = (unsigned short)f2bf_bits(f);
  __bf16 out;
  __builtin_memcpy(&out, &h, 2);
  return out;
}

// atomic max on float via signed/unsigned integer ordering trick.
// Valid when buffer initialized to -inf.
__device__ __forceinline__ void atomicMaxF32(float* addr, float val) {
  if (val >= 0.0f) atomicMax((int*)addr, __float_as_int(val));
  else             atomicMin((unsigned int*)addr, __float_as_uint(val));
}

// ---------------------------------------------------------------------------
// C[M x N] = A[M x K] (f32) * B[K x N] (f32), bf16 WMMA, f32 accumulate.
// Block = 256 threads = 8 waves covering 128 rows x 16 cols. The 32x16 B tile
// is staged per K-step in LDS *in WMMA fragment layout* (lane-major), so each
// lane fetches its 16 bf16 B elements as two aligned 16B vector loads.
// Fragment layout: element (kk, cc) lives at Bs[(cc + (kk & 16)) * 16 + (kk & 15)].
// ---------------------------------------------------------------------------
__global__ void gemm_bf16_wmma(const float* __restrict__ A, int lda,
                               const float* __restrict__ B, int ldb,
                               float* __restrict__ C, int ldc,
                               int K, int mblks) {
  __shared__ __align__(16) __bf16 Bs[32 * 16];
  const int tid  = threadIdx.x;
  const int wave = tid >> 5;
  const int lane = tid & 31;
  const int hl   = lane >> 4;   // lane half: selects K sub-pattern
  const int l15  = lane & 15;
  const int mb     = blockIdx.x % mblks;
  const int tile_n = blockIdx.x / mblks;
  const int tile_m = mb * 8 + wave;
  const int row  = tile_m * 16 + l15;    // A row for this lane
  const int colg = tile_n * 16 + l15;    // C col for this lane

  // cooperative B-load assignment: thread t loads rows kk, kk+1 at col cc
  const int cc = tid & 15;
  const int kk = (tid >> 4) * 2;
  const int bs_widx = (cc + (kk & 16)) * 8 + ((kk & 15) >> 1); // uint index

  v8f acc = {};
  for (int k0 = 0; k0 < K; k0 += 32) {
    // stage B tile: 2 f32 loads -> 1 packed bf16x2 LDS dword store per thread
    const float* bp = B + (size_t)(k0 + kk) * ldb + tile_n * 16 + cc;
    ((unsigned*)Bs)[bs_widx] = pack_bf2(bp[0], bp[ldb]);
    __syncthreads();

    // A fragment: lanes 0-15 hold K = 0..15, lanes 16-31 K = 16..31
    const float* ap = A + (size_t)row * lda + k0 + hl * 16;
    float4 g0 = *(const float4*)(ap + 0);
    float4 g1 = *(const float4*)(ap + 4);
    float4 g2 = *(const float4*)(ap + 8);
    float4 g3 = *(const float4*)(ap + 12);
    v8u au;
    au[0] = pack_bf2(g0.x, g0.y); au[1] = pack_bf2(g0.z, g0.w);
    au[2] = pack_bf2(g1.x, g1.y); au[3] = pack_bf2(g1.z, g1.w);
    au[4] = pack_bf2(g2.x, g2.y); au[5] = pack_bf2(g2.z, g2.w);
    au[6] = pack_bf2(g3.x, g3.y); au[7] = pack_bf2(g3.z, g3.w);
    const v16bf a = __builtin_bit_cast(v16bf, au);

    // B fragment: two 16B vector loads from lane-major LDS
    const v4u* bsrc = (const v4u*)(Bs + (l15 + hl * 16) * 16);
    const v4u b0 = bsrc[0], b1 = bsrc[1];
    const v16bf b = __builtin_bit_cast(
        v16bf, (v8u)__builtin_shufflevector(b0, b1, 0, 1, 2, 3, 4, 5, 6, 7));

    acc = __builtin_amdgcn_wmma_f32_16x16x32_bf16(false, a, false, b,
                                                  (short)0, acc, false, false);
    __syncthreads();
  }
#pragma unroll
  for (int r = 0; r < 8; ++r)
    C[(size_t)(tile_m * 16 + hl * 8 + r) * ldc + colg] = acc[r];
}

// ---------------------------------------------------------------------------
// Per-node attention logits: al_s[n,h] = dot(h[n,h,:], a_src[h,:]) (and dst)
// ---------------------------------------------------------------------------
__global__ void att_scores(const float* __restrict__ H,
                           const float* __restrict__ asrc,
                           const float* __restrict__ adst,
                           float* __restrict__ als, float* __restrict__ ald) {
  __shared__ float rs[CH], rd[CH];
  const int n = blockIdx.x, t = threadIdx.x;
  const float v = H[(size_t)n * CH + t];
  rs[t] = v * asrc[t];
  rd[t] = v * adst[t];
  __syncthreads();
  for (int off = 32; off >= 1; off >>= 1) {
    if ((t & 63) < off) { rs[t] += rs[t + off]; rd[t] += rd[t + off]; }
    __syncthreads();
  }
  if ((t & 63) == 0) {
    const int h = t >> 6;
    als[n * HEADS + h] = rs[t];
    ald[n * HEADS + h] = rd[t];
  }
}

__global__ void init_softmax(float* __restrict__ m, float* __restrict__ denom) {
  const int t = blockIdx.x * blockDim.x + threadIdx.x;
  if (t < NN * HEADS) { m[t] = -INFINITY; denom[t] = 0.0f; }
}

// e = leaky_relu(al_s[src] + al_d[dst]); store; segment max over dst
__global__ void edge_scores(const int* __restrict__ ei,
                            const float* __restrict__ als,
                            const float* __restrict__ ald,
                            float* __restrict__ esc, float* __restrict__ m) {
  const int t = blockIdx.x * blockDim.x + threadIdx.x;
  if (t >= EP4) return;
  const int e = t >> 2, h = t & 3;
  int s, d;
  if (e < EE) { s = ei[e]; d = ei[EE + e]; } else { s = d = e - EE; }
  float v = als[s * HEADS + h] + ald[d * HEADS + h];
  v = v > 0.0f ? v : 0.2f * v;            // leaky relu, slope 0.2
  esc[t] = v;
  atomicMaxF32(&m[d * HEADS + h], v);
}

// ex = exp(e - m[dst]); store over e; segment sum denominator
__global__ void edge_exp(const int* __restrict__ ei, float* __restrict__ esc,
                         const float* __restrict__ m, float* __restrict__ denom) {
  const int t = blockIdx.x * blockDim.x + threadIdx.x;
  if (t >= EP4) return;
  const int e = t >> 2, h = t & 3;
  const int d = (e < EE) ? ei[EE + e] : e - EE;
  float mm = m[d * HEADS + h];
  if (!(mm > -1e30f && mm < 1e30f)) mm = 0.0f;   // isfinite fixup
  const float ex = __expf(esc[t] - mm);
  esc[t] = ex;
  atomicAdd(&denom[d * HEADS + h], ex);
}

// agg[dst] += h[src] * alpha   (one wave per edge, 8 floats per lane)
__global__ void edge_aggregate(const int* __restrict__ ei,
                               const float* __restrict__ Hf,
                               const float* __restrict__ esc,
                               const float* __restrict__ denom,
                               float* __restrict__ agg) {
  const int e = blockIdx.x * 8 + (threadIdx.x >> 5);
  const int lane = threadIdx.x & 31;
  if (e >= EP) return;
  int s, d;
  if (e < EE) { s = ei[e]; d = ei[EE + e]; } else { s = d = e - EE; }
  const int h = lane >> 3;                      // 8 lanes per head
  const float alpha = esc[e * HEADS + h] / (denom[d * HEADS + h] + 1e-16f);
  const float4* hp = (const float4*)(Hf + (size_t)s * CH + lane * 8);
  const float4 g0 = hp[0], g1 = hp[1];
  float* op = agg + (size_t)d * CH + lane * 8;
  atomicAdd(op + 0, g0.x * alpha); atomicAdd(op + 1, g0.y * alpha);
  atomicAdd(op + 2, g0.z * alpha); atomicAdd(op + 3, g0.w * alpha);
  atomicAdd(op + 4, g1.x * alpha); atomicAdd(op + 5, g1.y * alpha);
  atomicAdd(op + 6, g1.z * alpha); atomicAdd(op + 7, g1.w * alpha);
}

__global__ void bias_relu(const float* __restrict__ agg,
                          const float* __restrict__ b, float* __restrict__ out) {
  const int t = blockIdx.x * 256 + threadIdx.x;
  out[t] = fmaxf(agg[t] + b[t & (CH - 1)], 0.0f);
}

// z = tanh(mean_heads(agg2) + b2); write fp32 to d_out tail + bf16 scratch copy
__global__ void finalize_z(const float* __restrict__ agg,
                           const float* __restrict__ b2,
                           float* __restrict__ zout, __bf16* __restrict__ zbf) {
  const int t = blockIdx.x * 256 + threadIdx.x;     // N*FOUT threads
  const int n = t >> 6, f = t & 63;
  const float* p = agg + (size_t)n * CH;
  const float sm = (p[f] + p[64 + f] + p[128 + f] + p[192 + f]) * 0.25f + b2[f];
  const float zv = tanhf(sm);
  zout[t] = zv;
  zbf[t]  = f2bf(zv);
}

// adj = sigmoid(z @ z^T), K = 64, bf16 WMMA. Each wave computes a 16x64 strip
// (4 N-tiles), reusing its A fragment 4x. B = z^T, so a lane's B fragment is
// 16 consecutive bf16 of a z row -> one 32B vector load.
__global__ void zzt_sigmoid_wmma(const __bf16* __restrict__ Z,
                                 float* __restrict__ out) {
  const int tid = threadIdx.x;
  const int wave = tid >> 5, lane = tid & 31, hl = lane >> 4, l15 = lane & 15;
  const int mblks = NN / 128;                       // 64
  const int mb  = blockIdx.x % mblks;
  const int tn4 = blockIdx.x / mblks;               // 64-col strip index
  const int tile_m = mb * 8 + wave;
  const int row  = tile_m * 16 + l15;
  const int col0 = tn4 * 64;

  v8f acc[4] = {};
#pragma unroll
  for (int k0 = 0; k0 < FOUT; k0 += 32) {
    const __bf16* apz = Z + (size_t)row * FOUT + k0 + hl * 16;
    const v4u a0 = *(const v4u*)apz;
    const v4u a1 = *(const v4u*)(apz + 8);
    const v16bf a = __builtin_bit_cast(
        v16bf, (v8u)__builtin_shufflevector(a0, a1, 0, 1, 2, 3, 4, 5, 6, 7));
#pragma unroll
    for (int j = 0; j < 4; ++j) {
      const v16bf b = __builtin_bit_cast(
          v16bf,
          *(const v8u*)(Z + (size_t)(col0 + j * 16 + l15) * FOUT + k0 + hl * 16));
      acc[j] = __builtin_amdgcn_wmma_f32_16x16x32_bf16(false, a, false, b,
                                                       (short)0, acc[j],
                                                       false, false);
    }
  }
#pragma unroll
  for (int r = 0; r < 8; ++r) {
    const size_t ro = (size_t)(tile_m * 16 + hl * 8 + r) * NN;
#pragma unroll
    for (int j = 0; j < 4; ++j)
      out[ro + col0 + j * 16 + l15] = 1.0f / (1.0f + __expf(-acc[j][r]));
  }
}

extern "C" void kernel_launch(void* const* d_in, const int* in_sizes, int n_in,
                              void* d_out, int out_size, void* d_ws, size_t ws_size,
                              hipStream_t stream) {
  (void)in_sizes; (void)n_in; (void)out_size; (void)ws_size;
  const float* x     = (const float*)d_in[0];
  const int*   ei    = (const int*)  d_in[1];
  const float* W1    = (const float*)d_in[2];
  const float* asrc1 = (const float*)d_in[3];
  const float* adst1 = (const float*)d_in[4];
  const float* b1    = (const float*)d_in[5];
  const float* W2    = (const float*)d_in[6];
  const float* asrc2 = (const float*)d_in[7];
  const float* adst2 = (const float*)d_in[8];
  const float* b2    = (const float*)d_in[9];
  float* out = (float*)d_out;

  // workspace layout (float units)
  float* ws    = (float*)d_ws;
  float* h1    = ws;                         // [N,256] h1, then h1act (reused)
  float* buf1  = ws + 2097152;               // [N,256] agg1, then h2 (reused)
  float* agg2  = ws + 4194304;               // [N,256]
  float* als   = ws + 6291456;               // [N,4]
  float* ald   = als + NN * HEADS;           // [N,4]
  float* mbuf  = ald + NN * HEADS;           // [N,4]
  float* den   = mbuf + NN * HEADS;          // [N,4]
  float* esc   = den + NN * HEADS;           // [EP,4]
  __bf16* zbf  = (__bf16*)(esc + EP4);       // [N,64] bf16

  // ---------------- layer 1 ----------------
  hipMemsetAsync(buf1, 0, (size_t)NN * CH * sizeof(float), stream);
  hipMemsetAsync(agg2, 0, (size_t)NN * CH * sizeof(float), stream);

  gemm_bf16_wmma<<<(NN / 128) * (CH / 16), 256, 0, stream>>>(
      x, NN, W1, CH, h1, CH, /*K=*/NN, /*mblks=*/NN / 128);
  att_scores<<<NN, 256, 0, stream>>>(h1, asrc1, adst1, als, ald);
  init_softmax<<<(NN * HEADS) / 256, 256, 0, stream>>>(mbuf, den);
  edge_scores<<<EP4 / 256, 256, 0, stream>>>(ei, als, ald, esc, mbuf);
  edge_exp<<<EP4 / 256, 256, 0, stream>>>(ei, esc, mbuf, den);
  edge_aggregate<<<EP / 8, 256, 0, stream>>>(ei, h1, esc, den, buf1);
  bias_relu<<<NN, 256, 0, stream>>>(buf1, b1, h1);       // h1 <- relu(agg1+b1)

  // ---------------- layer 2 ----------------
  gemm_bf16_wmma<<<(NN / 128) * (CH / 16), 256, 0, stream>>>(
      h1, CH, W2, CH, buf1, CH, /*K=*/CH, /*mblks=*/NN / 128);   // buf1 <- h2
  att_scores<<<NN, 256, 0, stream>>>(buf1, asrc2, adst2, als, ald);
  init_softmax<<<(NN * HEADS) / 256, 256, 0, stream>>>(mbuf, den);
  edge_scores<<<EP4 / 256, 256, 0, stream>>>(ei, als, ald, esc, mbuf);
  edge_exp<<<EP4 / 256, 256, 0, stream>>>(ei, esc, mbuf, den);
  edge_aggregate<<<EP / 8, 256, 0, stream>>>(ei, buf1, esc, den, agg2);

  // ---------------- decode ----------------
  finalize_z<<<(NN * FOUT) / 256, 256, 0, stream>>>(
      agg2, b2, out + (size_t)NN * NN, zbf);
  zzt_sigmoid_wmma<<<(NN / 128) * (NN / 64), 256, 0, stream>>>(zbf, out);
}
